// VanillaRNN_79345225827004
// MI455X (gfx1250) — matrix operand order repaired and checked by reading, not compiled, not run on hardware
//
#include <hip/hip_runtime.h>
#include <hip/hip_bf16.h>
#include <math.h>

#define BATCH 256
#define SEQ   256
#define INDIM 128
#define HID   1024
#define OUT_N 10
#define KTOT  (HID + INDIM)   // 1152 fused K (Whh rows + Whx rows)

#define NBLK    32            // persistent grid: 4 (M) x 8 (N) tiles
#define BM      64            // block tile M
#define BN      128           // block tile N
#define KC      32            // K chunk per WMMA stage
#define ASTRIDE (KC + 8)      // LDS row stride (elements), conflict-free padding
#define BSTRIDE (KC + 8)

typedef __attribute__((ext_vector_type(16))) __bf16 v16bf;
typedef __attribute__((ext_vector_type(8)))  float  v8f;
typedef unsigned short u16;

__device__ __forceinline__ u16 f2bf(float f) {
    union { float f; unsigned u; } v; v.f = f;
    unsigned u = v.u;
    return (u16)((u + 0x7FFFu + ((u >> 16) & 1u)) >> 16);
}
__device__ __forceinline__ float bf2f(u16 h) {
    union { unsigned u; float f; } v; v.u = ((unsigned)h) << 16;
    return v.f;
}

// ---- one-time conversions (recomputed every launch: deterministic) ----
__global__ void cvt_x_kernel(const float* __restrict__ x, u16* __restrict__ xb, int n) {
    int i = blockIdx.x * blockDim.x + threadIdx.x;
    if (i < n) xb[i] = f2bf(x[i]);
}

// Wt[n][k] (K-contiguous per output column): k<HID -> Whh[k][n], else Whx[k-HID][n]
__global__ void build_wt_kernel(const float* __restrict__ Whh, const float* __restrict__ Whx,
                                u16* __restrict__ Wt, int n_elems) {
    int i = blockIdx.x * blockDim.x + threadIdx.x;
    if (i >= n_elems) return;
    int n = i / KTOT;
    int k = i % KTOT;
    float v = (k < HID) ? Whh[k * HID + n] : Whx[(k - HID) * HID + n];
    Wt[i] = f2bf(v);
}

__global__ void zero_h_kernel(u16* __restrict__ h, int n) {
    int i = blockIdx.x * blockDim.x + threadIdx.x;
    if (i < n) h[i] = 0;
}

__global__ void zero_counter_kernel(int* __restrict__ c) {
    if (threadIdx.x == 0 && blockIdx.x == 0) *c = 0;
}

// ---- persistent recurrence: all 256 steps in ONE launch ----
// Step t: hnew = tanh([hprev | x_t] @ Wt^T + bh); device-wide barrier between steps.
union Frag { uint4 u[2]; v16bf v; };

__global__ __launch_bounds__(256) void rnn_persistent_kernel(
    const u16*  __restrict__ xb,   // [BATCH][SEQ][INDIM] bf16
    const u16*  __restrict__ Wt,   // [HID(n)][KTOT(k)] bf16, K contiguous
    const float* __restrict__ bh,  // [HID]
    u16* h0, u16* h1,              // ping-pong [BATCH][HID] bf16 (h0 pre-zeroed)
    int* counter)                  // grid-barrier counter (pre-zeroed)
{
    __shared__ __align__(16) u16 As[2][BM * ASTRIDE];  // 2 x 64x40 bf16 = 10 KB
    __shared__ __align__(16) u16 Bs[2][BN * BSTRIDE];  // 2 x 128x40 bf16 = 20 KB

    const int tid   = threadIdx.x;
    const int lane  = tid & 31;
    const int wave  = tid >> 5;          // 0..7, arranged 2(M) x 4(N)
    const int wm    = wave >> 2;
    const int wn    = wave & 3;
    const int lrow  = lane & 15;
    const int lhalf = lane >> 4;

    const int Mblk = (blockIdx.x & 3) * BM;   // 4 M tiles
    const int Nblk = (blockIdx.x >> 2) * BN;  // 8 N tiles
    const int M0 = Mblk + wm * 32;
    const int N0 = Nblk + wn * 32;

    // cooperative staging assignments (256 threads)
    const int a_row = tid >> 2;          // 0..63  : A tile 64 x 32, 1 uint4/thread
    const int a_col = (tid & 3) * 8;
    const int b_n   = tid >> 1;          // 0..127 : B tile 128 x 32, 2 uint4/thread
    const int b_k   = (tid & 1) * 16;

    for (int t = 0; t < SEQ; ++t) {
        const u16* hprev = (t & 1) ? h1 : h0;
        u16*       hnew  = (t & 1) ? h0 : h1;

        // prefetch chunk 0 (always from h since 0 < HID)
        uint4 ra, rb0, rb1;
        {
            ra = *(const uint4*)(hprev + (size_t)(Mblk + a_row) * HID + a_col);
            const u16* bsrc = Wt + (size_t)(Nblk + b_n) * KTOT + b_k;
            rb0 = *(const uint4*)(bsrc);
            rb1 = *(const uint4*)(bsrc + 8);
        }

        v8f acc[2][2] = {};

        for (int kc = 0, idx = 0; kc < KTOT; kc += KC, ++idx) {
            const int buf = idx & 1;
            // commit prefetched chunk to LDS
            *(uint4*)(&As[buf][a_row * ASTRIDE + a_col]) = ra;
            *(uint4*)(&Bs[buf][b_n * BSTRIDE + b_k])     = rb0;
            *(uint4*)(&Bs[buf][b_n * BSTRIDE + b_k + 8]) = rb1;
            __syncthreads();   // also protects buf^1 reuse two iterations apart

            // prefetch next chunk into registers (overlaps with WMMA below)
            const int kn = kc + KC;
            if (kn < KTOT) {
                if (kn < HID)
                    ra = *(const uint4*)(hprev + (size_t)(Mblk + a_row) * HID + kn + a_col);
                else
                    ra = *(const uint4*)(xb + ((size_t)(Mblk + a_row) * SEQ + t) * INDIM
                                         + (kn - HID) + a_col);
                const u16* bsrc = Wt + (size_t)(Nblk + b_n) * KTOT + kn + b_k;
                rb0 = *(const uint4*)(bsrc);
                rb1 = *(const uint4*)(bsrc + 8);
            }

            // WMMA fragments from LDS (documented 16-bit A/B lane layouts)
            Frag a[2], b[2];
#pragma unroll
            for (int i = 0; i < 2; ++i) {
                const int m = wm * 32 + i * 16 + lrow;
                a[i].u[0] = *(const uint4*)(&As[buf][m * ASTRIDE + 8 * lhalf]);
                a[i].u[1] = *(const uint4*)(&As[buf][m * ASTRIDE + 16 + 8 * lhalf]);
            }
#pragma unroll
            for (int j = 0; j < 2; ++j) {
                const int n = wn * 32 + j * 16 + lrow;
                b[j].u[0] = *(const uint4*)(&Bs[buf][n * BSTRIDE + 8 * lhalf]);
                b[j].u[1] = *(const uint4*)(&Bs[buf][n * BSTRIDE + 16 + 8 * lhalf]);
            }
#pragma unroll
            for (int i = 0; i < 2; ++i)
#pragma unroll
                for (int j = 0; j < 2; ++j)
                    acc[i][j] = __builtin_amdgcn_wmma_f32_16x16x32_bf16(
                        false, a[i].v, false, b[j].v, (short)0, acc[i][j], false, false);
        }

        // epilogue: + bias, tanh, store bf16 (C layout: M = r + 8*lhalf, N = lrow)
#pragma unroll
        for (int j = 0; j < 2; ++j) {
            const int n = N0 + j * 16 + lrow;
            const float bias = bh[n];
#pragma unroll
            for (int i = 0; i < 2; ++i)
#pragma unroll
                for (int r = 0; r < 8; ++r) {
                    const int m = M0 + i * 16 + r + 8 * lhalf;
                    hnew[(size_t)m * HID + n] = f2bf(tanhf(acc[i][j][r] + bias));
                }
        }

        // ---- device-wide barrier (release arrive, acquire spin) ----
        __syncthreads();
        if (tid == 0) {
            __hip_atomic_fetch_add(counter, 1, __ATOMIC_RELEASE, __HIP_MEMORY_SCOPE_AGENT);
            while (__hip_atomic_load(counter, __ATOMIC_ACQUIRE, __HIP_MEMORY_SCOPE_AGENT)
                   < NBLK * (t + 1)) {
                __builtin_amdgcn_s_sleep(2);
            }
        }
        __syncthreads();
    }
}

// ---- head: out = softmax(h_final @ Why + bo) ; tiny, one thread per batch row ----
__global__ void head_softmax_kernel(const u16* __restrict__ hfin,
                                    const float* __restrict__ Why,
                                    const float* __restrict__ bo,
                                    float* __restrict__ out)
{
    int b = blockIdx.x * blockDim.x + threadIdx.x;
    if (b >= BATCH) return;
    float o[OUT_N];
#pragma unroll
    for (int j = 0; j < OUT_N; ++j) o[j] = bo[j];
    for (int k = 0; k < HID; ++k) {
        float hv = bf2f(hfin[(size_t)b * HID + k]);
#pragma unroll
        for (int j = 0; j < OUT_N; ++j) o[j] += hv * Why[k * OUT_N + j];
    }
    float mx = o[0];
#pragma unroll
    for (int j = 1; j < OUT_N; ++j) mx = fmaxf(mx, o[j]);
    float s = 0.f;
#pragma unroll
    for (int j = 0; j < OUT_N; ++j) { o[j] = __expf(o[j] - mx); s += o[j]; }
    float inv = 1.0f / s;
#pragma unroll
    for (int j = 0; j < OUT_N; ++j) out[b * OUT_N + j] = o[j] * inv;
}

extern "C" void kernel_launch(void* const* d_in, const int* in_sizes, int n_in,
                              void* d_out, int out_size, void* d_ws, size_t ws_size,
                              hipStream_t stream) {
    (void)in_sizes; (void)n_in; (void)out_size; (void)ws_size;
    const float* x   = (const float*)d_in[0];
    const float* Whx = (const float*)d_in[1];
    const float* Whh = (const float*)d_in[2];
    const float* bh  = (const float*)d_in[3];
    const float* Why = (const float*)d_in[4];
    const float* bo  = (const float*)d_in[5];
    float* out = (float*)d_out;

    // workspace layout (bytes)
    char* ws = (char*)d_ws;
    const size_t xb_sz  = (size_t)BATCH * SEQ * INDIM * 2;  // 16,777,216
    const size_t wt_sz  = (size_t)HID * KTOT * 2;           //  2,359,296
    const size_t h_sz   = (size_t)BATCH * HID * 2;          //    524,288
    u16* xb  = (u16*)(ws);
    u16* Wt  = (u16*)(ws + xb_sz);
    u16* h0  = (u16*)(ws + xb_sz + wt_sz);
    u16* h1  = (u16*)(ws + xb_sz + wt_sz + h_sz);
    int* cnt = (int*)(ws + xb_sz + wt_sz + 2 * h_sz);

    const int nx = BATCH * SEQ * INDIM;
    cvt_x_kernel<<<(nx + 255) / 256, 256, 0, stream>>>(x, xb, nx);

    const int nw = HID * KTOT;
    build_wt_kernel<<<(nw + 255) / 256, 256, 0, stream>>>(Whh, Whx, Wt, nw);

    const int nh = BATCH * HID;
    zero_h_kernel<<<(nh + 255) / 256, 256, 0, stream>>>(h0, nh);
    zero_counter_kernel<<<1, 64, 0, stream>>>(cnt);

    // one persistent launch runs all 256 timesteps (32 blocks, all co-resident)
    rnn_persistent_kernel<<<NBLK, 256, 0, stream>>>(xb, Wt, bh, h0, h1, cnt);

    // after t=255 (odd), final hidden state is in h0
    head_softmax_kernel<<<(BATCH + 255) / 256, 256, 0, stream>>>(h0, Why, bo, out);
}
